// KANOriginal_82429012345259
// MI455X (gfx1250) — matrix operand order repaired and checked by reading, not compiled, not run on hardware
//
#include <hip/hip_runtime.h>

// ---------------------------------------------------------------------------
// KAN layer fused as one bf16 WMMA GEMM on gfx1250 (wave32).
// out[n,o] = sum_k A[n,k] * W[o,k] + bias[o],  K = Din*8 = 6144
//   A[n, i*8+0] = silu(x[n,i]);  A[n, i*8+1..6] = Bspline_{0..5}(x[n,i]);  slot7 = 0
//   W[o, i*8+0] = scale_base[o,i]; W[o, i*8+1..6] = scale_sp[o,i]*coef[o,i,m]; slot7 = 0
// W tile streamed global->LDS via GLOBAL_LOAD_ASYNC_TO_LDS_B128 (ASYNCcnt).
// ---------------------------------------------------------------------------

typedef __attribute__((ext_vector_type(16))) __bf16 bf16x16;
typedef __attribute__((ext_vector_type(8)))  float  v8f;

#define DIN   768
#define DOUT  768
#define NTOK  8192
#define KEXP  8              // expanded K-slots per input feature
#define KTOT  (DIN * KEXP)   // 6144
#define KCH   64             // K per LDS stage (8 input features)
#define KB    (KTOT / KCH)   // 96 stages
#define MT    32             // tokens per block
#define NBLK  256            // output columns per block
#define LDA   72             // LDS row stride (bf16 elems), padded vs 64 to spread banks
#define LDW   72

#if __has_builtin(__builtin_amdgcn_global_load_async_to_lds_b128)
#define USE_ASYNC_LDS 1
#else
#define USE_ASYNC_LDS 0
#endif

// Builtin signature: (v4i addrspace(1)* src, v4i addrspace(3)* dst, Ii off, Ii cpol)
typedef __attribute__((__vector_size__(16))) int v4i;
typedef __attribute__((address_space(1))) v4i* gv4p;  // global v4i*
typedef __attribute__((address_space(3))) v4i* lv4p;  // LDS v4i*

__device__ __forceinline__ void wait_async0() {
#if __has_builtin(__builtin_amdgcn_s_wait_asynccnt)
  __builtin_amdgcn_s_wait_asynccnt(0);
#else
  asm volatile("s_wait_asynccnt 0x0" ::: "memory");
#endif
}

__device__ __forceinline__ unsigned short f2bf(float f) {
  union { float f; unsigned u; } c; c.f = f;
  unsigned u = c.u;
  return (unsigned short)((u + 0x7FFFu + ((u >> 16) & 1u)) >> 16);  // RNE
}

// ---------------------------------------------------------------------------
// Kernel 1: pack weights into bf16 W[o][k] (row-major, k fastest) in d_ws.
// ---------------------------------------------------------------------------
__global__ void kan_pack_w(const float* __restrict__ coef,
                           const float* __restrict__ scale_base,
                           const float* __restrict__ scale_sp,
                           unsigned short* __restrict__ wp) {
  int idx = blockIdx.x * blockDim.x + threadIdx.x;
  if (idx >= DOUT * KTOT) return;
  int k = idx % KTOT;
  int o = idx / KTOT;
  int i = k >> 3;
  int s = k & 7;
  int oi = o * DIN + i;
  float v = 0.0f;
  if (s == 0)      v = scale_base[oi];
  else if (s <= 6) v = scale_sp[oi] * coef[oi * 6 + (s - 1)];
  wp[idx] = f2bf(v);
}

// ---------------------------------------------------------------------------
// Fragment loads per CDNA5 ISA 7.12.2 bf16 layouts (two ds_load_b128 each).
// A (16x32 MxK): lane m=l&15; lanes<16 hold K {h8..h8+7} U {16+h8..}, h8=(l>>4)*8
// B (32x16 KxN): lane n=l&15; holds K {h8..h8+7} U {16+h8..}, h8=(l>>4)*8 (same as A)
// ---------------------------------------------------------------------------
__device__ __forceinline__ bf16x16 frag_ld(const unsigned short* p0,
                                           const unsigned short* p1) {
  union { bf16x16 v; uint4 q[2]; } u;
  u.q[0] = *reinterpret_cast<const uint4*>(p0);
  u.q[1] = *reinterpret_cast<const uint4*>(p1);
  return u.v;
}

// ---------------------------------------------------------------------------
// Kernel 2: fused expand + GEMM + bias.
// Block: 256 threads = 8 waves. Tile: MT(32) tokens x NBLK(256) douts.
// Wave w owns 32x32: 2 M-frags x 2 N-frags -> 4 f32 accumulators.
// ---------------------------------------------------------------------------
__global__ __launch_bounds__(256, 2)
void kan_fused(const float* __restrict__ x,
               const unsigned short* __restrict__ wp,
               const float* __restrict__ bias,
               float* __restrict__ out) {
  __shared__ unsigned short aT[MT * LDA];    // 32 x 64 bf16 (+pad)  ~4.6 KB
  __shared__ unsigned short wT[NBLK * LDW];  // 256 x 64 bf16 (+pad) ~36.9 KB

  const int tid  = threadIdx.x;
  const int lane = tid & 31;
  const int wid  = tid >> 5;
  const int m0   = blockIdx.x * MT;
  const int nb0  = blockIdx.y * NBLK;

  const float h      = 2.0f / 3.0f;
  const float inv_h  = 1.0f / h;
  const float inv_2h = 1.0f / (2.0f * h);
  const float inv_3h = 1.0f / (3.0f * h);

  v8f acc[2][2];
#pragma unroll
  for (int a = 0; a < 2; ++a)
#pragma unroll
    for (int b = 0; b < 2; ++b) acc[a][b] = v8f{0.f,0.f,0.f,0.f,0.f,0.f,0.f,0.f};

  // expansion assignment: one x value per thread per stage
  const int et = tid >> 3;  // token row 0..31
  const int ef = tid & 7;   // feature within the 8-feature chunk
  unsigned short* aDst = &aT[et * LDA + ef * 8];

  const int fm = lane & 15;    // fragment lane row/col
  const int fh = lane >> 4;    // lane half

  for (int kb = 0; kb < KB; ++kb) {
    if (kb) __syncthreads();  // previous stage's LDS reads are done

    // ---- expand A tile: silu + cubic B-spline basis (Cox-de Boor) ----
    {
      float xv  = x[(m0 + et) * DIN + kb * 8 + ef];
      float sig = 1.0f / (1.0f + __expf(-xv));
      float silu = xv * sig;
      float b0[9];
#pragma unroll
      for (int t = 0; t < 9; ++t) {
        float gl = (float)t * h - 3.0f;
        b0[t] = (xv >= gl && xv < gl + h) ? 1.0f : 0.0f;
      }
      float b1[8];
#pragma unroll
      for (int j = 0; j < 8; ++j) {
        float gj = (float)j * h - 3.0f;
        b1[j] = (xv - gj) * inv_h * b0[j] + ((gj + 2.0f * h) - xv) * inv_h * b0[j + 1];
      }
      float b2[7];
#pragma unroll
      for (int j = 0; j < 7; ++j) {
        float gj = (float)j * h - 3.0f;
        b2[j] = (xv - gj) * inv_2h * b1[j] + ((gj + 3.0f * h) - xv) * inv_2h * b1[j + 1];
      }
      float b3[6];
#pragma unroll
      for (int j = 0; j < 6; ++j) {
        float gj = (float)j * h - 3.0f;
        b3[j] = (xv - gj) * inv_3h * b2[j] + ((gj + 4.0f * h) - xv) * inv_3h * b2[j + 1];
      }
      uint4 pk;
      pk.x = (unsigned)f2bf(silu)  | ((unsigned)f2bf(b3[0]) << 16);
      pk.y = (unsigned)f2bf(b3[1]) | ((unsigned)f2bf(b3[2]) << 16);
      pk.z = (unsigned)f2bf(b3[3]) | ((unsigned)f2bf(b3[4]) << 16);
      pk.w = (unsigned)f2bf(b3[5]);                 // slot 7 zero pad
      *reinterpret_cast<uint4*>(aDst) = pk;
    }

    // ---- stream W tile (256 rows x 64 bf16 = 32 KB) global -> LDS ----
#if USE_ASYNC_LDS
    {
      // Direct L2->LDS async copy, no VGPR staging (GLOBAL_LOAD_ASYNC_TO_LDS_B128,
      // tracked with ASYNCcnt). Thread tid copies its 128B row as 8 x 16B.
      const unsigned short* grow = wp + (size_t)(nb0 + tid) * KTOT + kb * KCH;
      unsigned short* lrow = &wT[tid * LDW];
#pragma unroll
      for (int c8 = 0; c8 < 8; ++c8) {
        __builtin_amdgcn_global_load_async_to_lds_b128(
            (gv4p)(grow + c8 * 8), (lv4p)(lrow + c8 * 8),
            /*offset=*/0, /*cpol=*/0);
      }
    }
#else
#pragma unroll
    for (int c = 0; c < 8; ++c) {
      int idx = c * 256 + tid;
      int row = idx >> 3;
      int c8  = idx & 7;
      const unsigned short* g = wp + (size_t)(nb0 + row) * KTOT + kb * KCH + c8 * 8;
      *reinterpret_cast<uint4*>(&wT[row * LDW + c8 * 8]) =
          *reinterpret_cast<const uint4*>(g);
    }
#endif
    // warm next W chunk (emits global_prefetch_b8); L2-resident weights
    if (kb + 1 < KB)
      __builtin_prefetch(wp + (size_t)(nb0 + tid) * KTOT + (kb + 1) * KCH, 0, 1);

#if USE_ASYNC_LDS
    wait_async0();  // this wave's async LDS writes complete; barrier covers rest
#endif
    __syncthreads();

    // ---- 2 K-steps of 32: 8 v_wmma_f32_16x16x32_bf16 per wave per stage ----
#pragma unroll
    for (int ks = 0; ks < 2; ++ks) {
      const unsigned short* a0p = &aT[(0 * 16 + fm) * LDA + ks * 32 + fh * 8];
      const unsigned short* a1p = &aT[(1 * 16 + fm) * LDA + ks * 32 + fh * 8];
      bf16x16 a0 = frag_ld(a0p, a0p + 16);
      bf16x16 a1 = frag_ld(a1p, a1p + 16);
#pragma unroll
      for (int j = 0; j < 2; ++j) {
        const unsigned short* bp =
            &wT[(wid * 32 + j * 16 + fm) * LDW + ks * 32 + fh * 8];
        bf16x16 bfr = frag_ld(bp, bp + 16);
        acc[0][j] = __builtin_amdgcn_wmma_f32_16x16x32_bf16(
            false, a0, false, bfr, (short)0, acc[0][j], false, false);
        acc[1][j] = __builtin_amdgcn_wmma_f32_16x16x32_bf16(
            false, a1, false, bfr, (short)0, acc[1][j], false, false);
      }
    }
  }

  // ---- store: C/D layout (VGPR r -> M = (lane>>4)*8 + r, N = lane&15) ----
#pragma unroll
  for (int am = 0; am < 2; ++am) {
#pragma unroll
    for (int j = 0; j < 2; ++j) {
      int o = nb0 + wid * 32 + j * 16 + fm;
      float bv = bias[o];
      int mrow = m0 + am * 16 + fh * 8;
#pragma unroll
      for (int r = 0; r < 8; ++r)
        out[(size_t)(mrow + r) * DOUT + o] = acc[am][j][r] + bv;
    }
  }
}

// ---------------------------------------------------------------------------
extern "C" void kernel_launch(void* const* d_in, const int* in_sizes, int n_in,
                              void* d_out, int out_size, void* d_ws, size_t ws_size,
                              hipStream_t stream) {
  const float* x          = (const float*)d_in[0];
  const float* coef       = (const float*)d_in[1];
  const float* scale_base = (const float*)d_in[2];
  const float* scale_sp   = (const float*)d_in[3];
  const float* bias       = (const float*)d_in[4];
  float* out              = (float*)d_out;
  unsigned short* wp      = (unsigned short*)d_ws;  // 768*6144*2 = 9.44 MB

  (void)in_sizes; (void)n_in; (void)out_size; (void)ws_size;

  kan_pack_w<<<(DOUT * KTOT + 255) / 256, 256, 0, stream>>>(coef, scale_base,
                                                            scale_sp, wp);
  dim3 grid(NTOK / MT, DOUT / NBLK);  // 256 x 3 blocks
  kan_fused<<<grid, 256, 0, stream>>>(x, wp, bias, out);
}
